// CausalHFPLayer_19851338842772
// MI455X (gfx1250) — matrix operand order repaired and checked by reading, not compiled, not run on hardware
//
#include <hip/hip_runtime.h>
#include <hip/hip_bf16.h>

// ---------------- problem constants ----------------
#define HID   1024
#define BSZ   4
#define NCH   64          // chunks per batch (nc)
#define CLEN  64          // chunk length C
#define MBINS 33          // rfft bins
#define KLOW  4           // low bins (attention path)
#define KHIGH 29          // high bins (conv path)
#define TLOW  256         // nc*KLOW tokens per batch
#define NHEAD 16
#define NKVH  4
#define HDIM  64
#define NCTOT 256         // B*nc

typedef __attribute__((ext_vector_type(16))) _Float16 v16h;
typedef __attribute__((ext_vector_type(8)))  _Float16 v8h;
typedef __attribute__((ext_vector_type(8)))  float    v8f;

// ---------------- fragment loaders (CDNA5 16-bit WMMA layouts) ----------------
__device__ __forceinline__ v16h load_a_frag(const _Float16* __restrict__ A, int lda,
                                            int m0, int k0, int lane) {
  int m  = m0 + (lane & 15);
  int kb = k0 + ((lane & 16) ? 16 : 0);
  const _Float16* p = A + (size_t)m * lda + kb;
  v8h lo = *(const v8h*)(p);        // K = base .. base+7
  v8h hi = *(const v8h*)(p + 8);    // K = base+8 .. base+15
  return __builtin_shufflevector(lo, hi, 0,1,2,3,4,5,6,7,8,9,10,11,12,13,14,15);
}

__device__ __forceinline__ v16h load_b_frag(const _Float16* __restrict__ W, int ldw,
                                            int n0, int k0, int lane) {
  int n  = n0 + (lane & 15);
  int kb = k0 + ((lane & 16) ? 16 : 0);
  return *(const v16h*)(W + (size_t)n * ldw + kb);   // 16 contiguous K values
}

// ---------------- generic GEMM: C[M,N] = A[M,K] * W[N,K]^T ----------------
// grid = (N/256, M/64), block = 128 (4 waves); each wave does a 64x64 square
// register tile (4 A frags x 4 B frags = 16 WMMAs per k-step; 32 flops/byte).
__global__ __launch_bounds__(128) void gemm_f16_wmma(const _Float16* __restrict__ A,
                                                     const _Float16* __restrict__ W,
                                                     float* __restrict__ C,
                                                     int M, int N, int K) {
  const int lane = threadIdx.x & 31;
  const int wave = threadIdx.x >> 5;
  const int m0   = blockIdx.y * 64;
  const int n0   = blockIdx.x * 256 + wave * 64;
  if (m0 >= M || n0 >= N) return;

  v8f acc[4][4] = {};
  for (int k0 = 0; k0 < K; k0 += 32) {
    v16h a[4];
#pragma unroll
    for (int i = 0; i < 4; ++i) a[i] = load_a_frag(A, K, m0 + i * 16, k0, lane);
#pragma unroll
    for (int t = 0; t < 4; ++t) {
      v16h b = load_b_frag(W, K, n0 + t * 16, k0, lane);
#pragma unroll
      for (int i = 0; i < 4; ++i)
        acc[i][t] = __builtin_amdgcn_wmma_f32_16x16x32_f16(false, a[i], false, b,
                                                           (short)0, acc[i][t],
                                                           false, false);
    }
  }
  const int nCol = lane & 15;
  const int mOff = (lane & 16) ? 8 : 0;
#pragma unroll
  for (int i = 0; i < 4; ++i) {
#pragma unroll
    for (int t = 0; t < 4; ++t) {
      float* cp = C + (size_t)(m0 + i * 16 + mOff) * N + (n0 + t * 16 + nCol);
#pragma unroll
      for (int r = 0; r < 8; ++r) cp[(size_t)r * N] = acc[i][t][r];
    }
  }
}

// ---------------- fp32 -> fp16 convert ----------------
__global__ void f32_to_f16(const float* __restrict__ s, _Float16* __restrict__ d, int n) {
  int i = blockIdx.x * blockDim.x + threadIdx.x;
  if (i < n) d[i] = (_Float16)s[i];
}

// ---------------- forward DFT (rfft over C=64) ----------------
// grid = 256 (b*nc), block = 256; each thread handles 4 feature columns.
__global__ __launch_bounds__(256) void dft_kernel(const float* __restrict__ hs,
    float* __restrict__ XLr, float* __restrict__ XLi,
    _Float16* __restrict__ XLrh, _Float16* __restrict__ XLih,
    float* __restrict__ XHr, float* __restrict__ XHi,
    _Float16* __restrict__ meanh) {
  const int bc = blockIdx.x;           // b*64 + c
  const int b = bc >> 6, c = bc & 63;
  __shared__ float ct[64], st[64];
  if (threadIdx.x < 64) {
    float ang = -6.28318530717958647692f * (float)threadIdx.x / 64.0f;
    ct[threadIdx.x] = cosf(ang);
    st[threadIdx.x] = sinf(ang);
  }
  __syncthreads();

  for (int col = 0; col < 4; ++col) {
    const int d = threadIdx.x + col * 256;
    const float* xp = hs + ((size_t)b * 4096 + (size_t)c * 64) * HID + d;
    float x[64];
#pragma unroll
    for (int n = 0; n < 64; ++n) x[n] = xp[(size_t)n * HID];

    for (int m = 0; m < MBINS; ++m) {
      const float cm = ct[m], sm = st[m];     // e^{-2*pi*i*m/64}
      float cr = 1.f, ci = 0.f;
      float ar = 0.f, ai = 0.f;
#pragma unroll
      for (int n = 0; n < 64; ++n) {
        ar += x[n] * cr;
        ai += x[n] * ci;
        float t = cr * cm - ci * sm;
        ci = cr * sm + ci * cm;
        cr = t;
      }
      if (m < KLOW) {
        size_t li = ((size_t)(b * TLOW + c * KLOW + m)) * HID + d;
        XLr[li] = ar;  XLi[li] = ai;
        XLrh[li] = (_Float16)ar;  XLih[li] = (_Float16)ai;
        if (m == 0) meanh[(size_t)bc * HID + d] = (_Float16)(ar * 0.015625f);
      } else {
        size_t hi = ((size_t)bc * KHIGH + (m - KLOW)) * HID + d;
        XHr[hi] = ar;  XHi[hi] = ai;
      }
    }
  }
}

// ---------------- block-causal GQA attention (one (b,head) per block) ----------------
// Q: (B,T,1024) fp32; Kb/Vb: (B,T,256) fp32; Oh: (B,T,1024) fp16
__global__ __launch_bounds__(256) void attn_kernel(const float* __restrict__ Q,
                                                   const float* __restrict__ Kb,
                                                   const float* __restrict__ Vb,
                                                   _Float16* __restrict__ Oh) {
  const int bh = blockIdx.x;           // b*16 + h
  const int b = bh >> 4, h = bh & 15;
  const int kvh = h >> 2;              // repeat factor 4
  __shared__ _Float16 Ks[TLOW * HDIM];
  __shared__ _Float16 Vs[TLOW * HDIM];
  for (int i = threadIdx.x; i < TLOW * HDIM; i += 256) {
    int t = i >> 6, j = i & 63;
    size_t src = ((size_t)b * TLOW + t) * (NKVH * HDIM) + kvh * HDIM + j;
    Ks[i] = (_Float16)Kb[src];
    Vs[i] = (_Float16)Vb[src];
  }
  __syncthreads();

  const int t = threadIdx.x;           // query row
  float q[64];
  const float* qp = Q + ((size_t)b * TLOW + t) * HID + h * HDIM;
#pragma unroll
  for (int j = 0; j < 64; ++j) q[j] = qp[j] * 0.125f;   // 1/sqrt(64)

  float mx = -3.402823466e38f, l = 0.f, acc[64];
#pragma unroll
  for (int j = 0; j < 64; ++j) acc[j] = 0.f;

  const int kmax = ((t >> 2) + 1) << 2;  // block-causal: keys with chunk <= qchunk
  for (int kk = 0; kk < kmax; ++kk) {
    float s = 0.f;
    const _Float16* kp = Ks + kk * HDIM;
#pragma unroll
    for (int j = 0; j < 64; ++j) s += q[j] * (float)kp[j];
    float mn   = fmaxf(mx, s);
    float corr = expf(mx - mn);
    float p    = expf(s - mn);
    l = l * corr + p;
    const _Float16* vp = Vs + kk * HDIM;
#pragma unroll
    for (int j = 0; j < 64; ++j) acc[j] = acc[j] * corr + p * (float)vp[j];
    mx = mn;
  }
  _Float16* op = Oh + ((size_t)b * TLOW + t) * HID + h * HDIM;
  float inv = 1.f / l;
#pragma unroll
  for (int j = 0; j < 64; ++j) op[j] = (_Float16)(acc[j] * inv);
}

// ---------------- depthwise causal conv (KS=7) + exact GELU -> f16 ----------------
__global__ void dwconv_gelu(const float* __restrict__ X, const float* __restrict__ dw,
                            _Float16* __restrict__ G, int total) {
  int idx = blockIdx.x * blockDim.x + threadIdx.x;
  if (idx >= total) return;
  int d = idx & (HID - 1);
  int l = (idx >> 10) % KHIGH;
  int n = idx / (KHIGH * HID);
  const float* xp = X + (size_t)n * KHIGH * HID + d;
  float acc = 0.f;
#pragma unroll
  for (int j = 0; j < 7; ++j) {
    int li = l - 6 + j;
    if (li >= 0) acc += xp[(size_t)li * HID] * dw[d * 7 + j];
  }
  float g = 0.5f * acc * (1.f + erff(acc * 0.70710678118654752f));  // exact gelu
  G[idx] = (_Float16)g;
}

// ---------------- blend + inverse rfft + write output ----------------
__global__ __launch_bounds__(256) void final_kernel(
    const float* __restrict__ XLr, const float* __restrict__ XLi,
    const float* __restrict__ OPr, const float* __restrict__ OPi,
    const float* __restrict__ XHr, const float* __restrict__ XHi,
    const float* __restrict__ Hr,  const float* __restrict__ Hi,
    const float* __restrict__ Gate, const float* __restrict__ alpha_raw,
    float* __restrict__ out) {
  const int bc = blockIdx.x;           // b*64 + c
  const int b = bc >> 6, c = bc & 63;
  __shared__ float ct[64], st[64];
  if (threadIdx.x < 64) {
    float ang = 6.28318530717958647692f * (float)threadIdx.x / 64.0f;
    ct[threadIdx.x] = cosf(ang);
    st[threadIdx.x] = sinf(ang);
  }
  __syncthreads();
  const float alpha = 1.f / (1.f + expf(-alpha_raw[0]));

  for (int col = 0; col < 4; ++col) {
    const int d = threadIdx.x + col * 256;
    float re[MBINS], im[MBINS];
#pragma unroll
    for (int m = 0; m < KLOW; ++m) {
      size_t li = ((size_t)(b * TLOW + c * KLOW + m)) * HID + d;
      re[m] = alpha * OPr[li] + (1.f - alpha) * XLr[li];
      im[m] = alpha * OPi[li] + (1.f - alpha) * XLi[li];
    }
    const float g = 1.f / (1.f + expf(-Gate[(size_t)bc * HID + d]));
#pragma unroll
    for (int m = KLOW; m < MBINS; ++m) {
      size_t hi = ((size_t)bc * KHIGH + (m - KLOW)) * HID + d;
      re[m] = Hr[hi] * g + XHr[hi];
      im[m] = Hi[hi] * g + XHi[hi];
    }
    for (int n = 0; n < 64; ++n) {
      float cn = ct[n], sn = st[n];    // e^{+2*pi*i*n/64}
      float cr = cn, ci = sn;          // value at m=1
      float s = re[0] + ((n & 1) ? -re[32] : re[32]);
#pragma unroll
      for (int m = 1; m < 32; ++m) {
        s += 2.f * (re[m] * cr - im[m] * ci);
        float t2 = cr * cn - ci * sn;
        ci = cr * sn + ci * cn;
        cr = t2;
      }
      out[((size_t)b * 4096 + (size_t)c * 64 + n) * HID + d] = s * 0.015625f;
    }
  }
}

// ---------------- host-side orchestration ----------------
extern "C" void kernel_launch(void* const* d_in, const int* in_sizes, int n_in,
                              void* d_out, int out_size, void* d_ws, size_t ws_size,
                              hipStream_t stream) {
  const float* hs     = (const float*)d_in[0];
  const float* q_w    = (const float*)d_in[1];
  const float* k_w    = (const float*)d_in[2];
  const float* v_w    = (const float*)d_in[3];
  const float* o_w    = (const float*)d_in[4];
  const float* dw_r   = (const float*)d_in[5];
  const float* pw_r   = (const float*)d_in[6];
  const float* dw_i   = (const float*)d_in[7];
  const float* pw_i   = (const float*)d_in[8];
  const float* gate_w = (const float*)d_in[9];
  const float* a_raw  = (const float*)d_in[10];
  float* out = (float*)d_out;

  const size_t NLOW  = (size_t)BSZ * TLOW * HID;       // 1,048,576
  const size_t NHIGH = (size_t)NCTOT * KHIGH * HID;    // 7,602,176
  const size_t NKV   = (size_t)BSZ * TLOW * (NKVH*HDIM);

  char* p = (char*)d_ws;
  auto alloc = [&](size_t bytes) -> void* {
    void* r = (void*)p;
    p += (bytes + 255) & ~(size_t)255;
    return r;
  };
  float*    XLr   = (float*)alloc(NLOW * 4);
  float*    XLi   = (float*)alloc(NLOW * 4);
  _Float16* XLrh  = (_Float16*)alloc(NLOW * 2);
  _Float16* XLih  = (_Float16*)alloc(NLOW * 2);
  float*    XHr   = (float*)alloc(NHIGH * 4);
  float*    XHi   = (float*)alloc(NHIGH * 4);
  _Float16* meanh = (_Float16*)alloc((size_t)NCTOT * HID * 2);
  _Float16* qwh   = (_Float16*)alloc((size_t)HID * HID * 2);
  _Float16* kwh   = (_Float16*)alloc((size_t)(NKVH*HDIM) * HID * 2);
  _Float16* vwh   = (_Float16*)alloc((size_t)(NKVH*HDIM) * HID * 2);
  _Float16* owh   = (_Float16*)alloc((size_t)HID * HID * 2);
  _Float16* pwrh  = (_Float16*)alloc((size_t)HID * HID * 2);
  _Float16* pwih  = (_Float16*)alloc((size_t)HID * HID * 2);
  _Float16* gwh   = (_Float16*)alloc((size_t)HID * HID * 2);
  float*    Qr    = (float*)alloc(NLOW * 4);
  float*    Qi    = (float*)alloc(NLOW * 4);
  float*    Kr    = (float*)alloc(NKV * 4);
  float*    Ki    = (float*)alloc(NKV * 4);
  float*    Vr    = (float*)alloc(NKV * 4);
  float*    Vi    = (float*)alloc(NKV * 4);
  _Float16* Ohr   = (_Float16*)alloc(NLOW * 2);
  _Float16* Ohi   = (_Float16*)alloc(NLOW * 2);
  float*    OPr   = (float*)alloc(NLOW * 4);
  float*    OPi   = (float*)alloc(NLOW * 4);
  _Float16* Grh   = (_Float16*)alloc(NHIGH * 2);
  _Float16* Gih   = (_Float16*)alloc(NHIGH * 2);
  float*    Hr    = (float*)alloc(NHIGH * 4);
  float*    Hi    = (float*)alloc(NHIGH * 4);
  float*    GateO = (float*)alloc((size_t)NCTOT * HID * 4);

  // 1) weight conversions to f16
  const int WFULL = HID * HID, WKV = (NKVH*HDIM) * HID;
  f32_to_f16<<<dim3((WFULL + 255) / 256), dim3(256), 0, stream>>>(q_w,    qwh,  WFULL);
  f32_to_f16<<<dim3((WKV   + 255) / 256), dim3(256), 0, stream>>>(k_w,    kwh,  WKV);
  f32_to_f16<<<dim3((WKV   + 255) / 256), dim3(256), 0, stream>>>(v_w,    vwh,  WKV);
  f32_to_f16<<<dim3((WFULL + 255) / 256), dim3(256), 0, stream>>>(o_w,    owh,  WFULL);
  f32_to_f16<<<dim3((WFULL + 255) / 256), dim3(256), 0, stream>>>(pw_r,   pwrh, WFULL);
  f32_to_f16<<<dim3((WFULL + 255) / 256), dim3(256), 0, stream>>>(pw_i,   pwih, WFULL);
  f32_to_f16<<<dim3((WFULL + 255) / 256), dim3(256), 0, stream>>>(gate_w, gwh,  WFULL);

  // 2) forward DFT (splits low/high bins, f16 copies, chunk means)
  dft_kernel<<<dim3(NCTOT), dim3(256), 0, stream>>>(hs, XLr, XLi, XLrh, XLih,
                                                    XHr, XHi, meanh);

  // 3) Q/K/V projections (WMMA GEMMs): M = B*T = 1024, K = 1024
  gemm_f16_wmma<<<dim3(4, 16), dim3(128), 0, stream>>>(XLrh, qwh, Qr, 1024, 1024, 1024);
  gemm_f16_wmma<<<dim3(4, 16), dim3(128), 0, stream>>>(XLih, qwh, Qi, 1024, 1024, 1024);
  gemm_f16_wmma<<<dim3(1, 16), dim3(128), 0, stream>>>(XLrh, kwh, Kr, 1024,  256, 1024);
  gemm_f16_wmma<<<dim3(1, 16), dim3(128), 0, stream>>>(XLih, kwh, Ki, 1024,  256, 1024);
  gemm_f16_wmma<<<dim3(1, 16), dim3(128), 0, stream>>>(XLrh, vwh, Vr, 1024,  256, 1024);
  gemm_f16_wmma<<<dim3(1, 16), dim3(128), 0, stream>>>(XLih, vwh, Vi, 1024,  256, 1024);

  // 4) block-causal GQA attention, real & imag
  attn_kernel<<<dim3(BSZ * NHEAD), dim3(256), 0, stream>>>(Qr, Kr, Vr, Ohr);
  attn_kernel<<<dim3(BSZ * NHEAD), dim3(256), 0, stream>>>(Qi, Ki, Vi, Ohi);

  // 5) O-projection (WMMA)
  gemm_f16_wmma<<<dim3(4, 16), dim3(128), 0, stream>>>(Ohr, owh, OPr, 1024, 1024, 1024);
  gemm_f16_wmma<<<dim3(4, 16), dim3(128), 0, stream>>>(Ohi, owh, OPi, 1024, 1024, 1024);

  // 6) depthwise conv + GELU (high bins)
  const int CONVTOT = (int)NHIGH;
  dwconv_gelu<<<dim3((CONVTOT + 255) / 256), dim3(256), 0, stream>>>(XHr, dw_r, Grh, CONVTOT);
  dwconv_gelu<<<dim3((CONVTOT + 255) / 256), dim3(256), 0, stream>>>(XHi, dw_i, Gih, CONVTOT);

  // 7) pointwise projections (dominant WMMA GEMMs): M = 7424 = 116*64
  gemm_f16_wmma<<<dim3(4, 116), dim3(128), 0, stream>>>(Grh, pwrh, Hr, 7424, 1024, 1024);
  gemm_f16_wmma<<<dim3(4, 116), dim3(128), 0, stream>>>(Gih, pwih, Hi, 7424, 1024, 1024);

  // 8) gate GEMM: M = B*nc = 256
  gemm_f16_wmma<<<dim3(4, 4), dim3(128), 0, stream>>>(meanh, gwh, GateO, 256, 1024, 1024);

  // 9) blend + inverse rfft + output
  final_kernel<<<dim3(NCTOT), dim3(256), 0, stream>>>(XLr, XLi, OPr, OPi, XHr, XHi,
                                                      Hr, Hi, GateO, a_raw, out);
  (void)in_sizes; (void)n_in; (void)out_size; (void)ws_size;
}
